// DistributedMoE_70446053589285
// MI455X (gfx1250) — compile-verified, matrix operand order checked by reading, not run on hardware
//
#include <hip/hip_runtime.h>
#include <hip/hip_bf16.h>
#include <math.h>

typedef __bf16 bf16_t;
typedef __attribute__((ext_vector_type(16))) __bf16 v16bf;
typedef __attribute__((ext_vector_type(8)))  __bf16 v8bf;
typedef __attribute__((ext_vector_type(4)))  __bf16 v4bf;
typedef __attribute__((ext_vector_type(8)))  float  v8f;

typedef __attribute__((ext_vector_type(4))) unsigned int u32x4;
typedef __attribute__((ext_vector_type(8))) int          i32x8;
typedef __attribute__((ext_vector_type(4))) int          i32x4;

#define TILE_M 128
#define TILE_N 128
#define TILE_K 32
#define LDK    48                          // padded LDS row: 64B data + 32B pad = 96B
#define TILE_BYTES (TILE_M * LDK * 2)      // 12288 B per staged tile
#define SMEM_BYTES (4 * TILE_BYTES)        // A0,B0,A1,B1 double buffer = 48 KB

#if __has_builtin(__builtin_amdgcn_tensor_load_to_lds) && \
    __has_builtin(__builtin_amdgcn_s_wait_tensorcnt)
#define USE_TDM 1
#else
#define USE_TDM 0
#endif

// ---------------- conversion kernels ----------------

__global__ __launch_bounds__(256) void cvt_f32_bf16(const float* __restrict__ src,
                                                    bf16_t* __restrict__ dst, int n4) {
  int i = blockIdx.x * blockDim.x + threadIdx.x;
  if (i >= n4) return;
  const float4 v = ((const float4*)src)[i];
  v4bf o;
  o.x = (bf16_t)v.x; o.y = (bf16_t)v.y; o.z = (bf16_t)v.z; o.w = (bf16_t)v.w;
  ((v4bf*)dst)[i] = o;
}

// src[R][C] fp32 (row-major) -> dst[C][R] bf16 (row-major)
__global__ __launch_bounds__(256) void transpose_cvt(const float* __restrict__ src,
                                                     bf16_t* __restrict__ dst,
                                                     int R, int C) {
  __shared__ float tile[32][33];
  const int tx = threadIdx.x, ty = threadIdx.y;
  const int c  = blockIdx.x * 32 + tx;
  const int r0 = blockIdx.y * 32;
  #pragma unroll
  for (int j = ty; j < 32; j += 8)
    tile[j][tx] = src[(size_t)(r0 + j) * C + c];
  __syncthreads();
  const int r  = r0 + tx;
  const int cb = blockIdx.x * 32;
  #pragma unroll
  for (int j = ty; j < 32; j += 8)
    dst[(size_t)(cb + j) * R + r] = (bf16_t)tile[tx][j];
}

// ---------------- router: p7[t] = softmax(x[t] @ Wr^T)[7] ----------------

__global__ __launch_bounds__(256) void router_p7(const float* __restrict__ x,
                                                 const float* __restrict__ Wr,
                                                 float* __restrict__ p7, int H) {
  const int lane = threadIdx.x & 31;
  const int tok  = (blockIdx.x * blockDim.x + threadIdx.x) >> 5;
  const float* xr = x + (size_t)tok * H;
  float acc[8] = {0.f,0.f,0.f,0.f,0.f,0.f,0.f,0.f};
  for (int k = lane * 4; k < H; k += 128) {
    const float4 xv = *(const float4*)(xr + k);
    #pragma unroll
    for (int e = 0; e < 8; ++e) {
      const float4 wv = *(const float4*)(Wr + (size_t)e * H + k);
      acc[e] += xv.x * wv.x + xv.y * wv.y + xv.z * wv.z + xv.w * wv.w;
    }
  }
  #pragma unroll
  for (int e = 0; e < 8; ++e) {
    #pragma unroll
    for (int off = 16; off > 0; off >>= 1)
      acc[e] += __shfl_xor(acc[e], off, 32);
  }
  if (lane == 0) {
    float mx = acc[0];
    #pragma unroll
    for (int e = 1; e < 8; ++e) mx = fmaxf(mx, acc[e]);
    float s = 0.f;
    #pragma unroll
    for (int e = 0; e < 8; ++e) s += expf(acc[e] - mx);
    p7[tok] = expf(acc[7] - mx) / s;
  }
}

// ---------------- TDM tile load ----------------
// Stage a TILE_M x TILE_K bf16 tile (row stride K elems in memory) into LDS at
// byte offset ldsOff, with hardware row padding 64B->96B (pad_interval=16 DW,
// pad_amount=8 DW) so LDS rows land on LDK=48-element stride.
#if USE_TDM
__device__ __forceinline__ void tdm_tile_load(const bf16_t* gsrc, unsigned ldsOff, int K) {
  const unsigned long long ga = (unsigned long long)(const void*)gsrc;
  u32x4 g0;
  g0[0] = 1u;                                                  // count=1 (valid), user mode
  g0[1] = ldsOff;                                              // lds_addr (bytes)
  g0[2] = (unsigned)ga;                                        // global_addr[31:0]
  g0[3] = (unsigned)((ga >> 32) & 0x01FFFFFFu) | (2u << 30);   // addr[56:32] | type=2
  i32x8 g1;
  const unsigned td0 = (unsigned)K;        // tensor_dim0 (elements)
  const unsigned td1 = 1u << 20;           // tensor_dim1: large, no OOB
  g1[0] = (int)((1u << 16)                 // data_size = 2 bytes
              | (1u << 20)                 // pad_enable
              | (3u << 22)                 // pad_interval: 16 DWORDs (64B)
              | (7u << 25));               // pad_amount:   8 DWORDs (32B)
  g1[1] = (int)((td0 & 0xFFFFu) << 16);                        // [63:48]=td0.lo
  g1[2] = (int)((td0 >> 16) | ((td1 & 0xFFFFu) << 16));        // td0.hi | td1.lo
  g1[3] = (int)((td1 >> 16) | ((unsigned)TILE_K << 16));       // td1.hi | tile_dim0
  g1[4] = (int)TILE_M;                                         // tile_dim1 (tile_dim2=0)
  g1[5] = (int)K;                                              // tensor_dim0_stride lo
  g1[6] = 0;
  g1[7] = 0;
  const i32x4 gz4 = {0, 0, 0, 0};
  const i32x8 gz8 = {0, 0, 0, 0, 0, 0, 0, 0};
  // 6-arg form (clang-23 / therock-10.0 headers): g0, g1, g2, g3, extra, cpol
  __builtin_amdgcn_tensor_load_to_lds(g0, g1, gz4, gz4, gz8, 0);
}
#endif

// ---------------- WMMA GEMM mainloop ----------------
// A: [M][K] bf16 row-major, Bt: [N][K] bf16 row-major (B pre-transposed).
// 8 waves; wave (wm,wn) owns a 64x32 sub-tile = 4x2 fragments of 16x16.
__device__ __forceinline__ void gemm_tiles(const bf16_t* __restrict__ A,
                                           const bf16_t* __restrict__ Bt,
                                           int K, int mBlk, int nBlk,
                                           v8f acc[4][2]) {
  extern __shared__ __align__(16) char smem[];
  bf16_t* const lds = (bf16_t*)smem;

  const int tid   = threadIdx.x;
  const int lane  = tid & 31;
  const int wid   = tid >> 5;
  const int waveM = wid & 1;
  const int waveN = wid >> 1;
  const int l16   = lane & 15;
  const int c0    = (lane >> 4) * 8;

  const bf16_t* gA = A  + (size_t)mBlk * TILE_M * K;
  const bf16_t* gB = Bt + (size_t)nBlk * TILE_N * K;
  const int nk = K / TILE_K;
  const bool issuer = (wid == 0);

#if USE_TDM
  if (issuer) {
    tdm_tile_load(gA, 0u, K);
    tdm_tile_load(gB, (unsigned)TILE_BYTES, K);
  }
#else
  const int rowL = tid >> 2;          // 0..63
  const int colL = (tid & 3) * 8;     // 0,8,16,24
#endif

  for (int i = 0; i < nk; ++i) {
    const int cur = i & 1;
    bf16_t* const As = lds + cur * (2 * TILE_M * LDK);
    bf16_t* const Bs = As + TILE_M * LDK;

#if USE_TDM
    if (issuer) __builtin_amdgcn_s_wait_tensorcnt(0);
    __syncthreads();   // buffer `cur` ready; also: all waves done reading buffer !cur
    if (issuer && (i + 1 < nk)) {
      const unsigned nb = (unsigned)((cur ^ 1) * 2 * TILE_BYTES);
      tdm_tile_load(gA + (size_t)(i + 1) * TILE_K, nb, K);
      tdm_tile_load(gB + (size_t)(i + 1) * TILE_K, nb + (unsigned)TILE_BYTES, K);
    }
#else
    const bf16_t* ga = gA + (size_t)rowL * K + (size_t)i * TILE_K + colL;
    const bf16_t* gb = gB + (size_t)rowL * K + (size_t)i * TILE_K + colL;
    const size_t rstride = (size_t)64 * K;
    __syncthreads();
    *(v8bf*)(As + rowL * LDK + colL)        = *(const v8bf*)(ga);
    *(v8bf*)(As + (rowL + 64) * LDK + colL) = *(const v8bf*)(ga + rstride);
    *(v8bf*)(Bs + rowL * LDK + colL)        = *(const v8bf*)(gb);
    *(v8bf*)(Bs + (rowL + 64) * LDK + colL) = *(const v8bf*)(gb + rstride);
    __builtin_prefetch(ga + TILE_K, 0, 1);
    __builtin_prefetch(gb + TILE_K, 0, 1);
    __syncthreads();
#endif

    v16bf bfrag[2];
    #pragma unroll
    for (int fn = 0; fn < 2; ++fn) {
      const bf16_t* p = Bs + (waveN * 32 + fn * 16 + l16) * LDK;
      v8bf lo = *(const v8bf*)(p + c0);
      v8bf hi = *(const v8bf*)(p + 16 + c0);
      bfrag[fn] = __builtin_shufflevector(lo, hi, 0,1,2,3,4,5,6,7,8,9,10,11,12,13,14,15);
    }
    #pragma unroll
    for (int fm = 0; fm < 4; ++fm) {
      const bf16_t* p = As + (waveM * 64 + fm * 16 + l16) * LDK;
      v8bf lo = *(const v8bf*)(p + c0);
      v8bf hi = *(const v8bf*)(p + 16 + c0);
      v16bf afrag = __builtin_shufflevector(lo, hi, 0,1,2,3,4,5,6,7,8,9,10,11,12,13,14,15);
      #pragma unroll
      for (int fn = 0; fn < 2; ++fn)
        acc[fm][fn] = __builtin_amdgcn_wmma_f32_16x16x32_bf16(
            false, afrag, false, bfrag[fn], (short)0, acc[fm][fn], false, false);
    }
  }
}

// GEMM1: H = gelu(X @ W1 + b1), output bf16 [T][F]
__global__ __launch_bounds__(256) void gemm1_gelu(const bf16_t* __restrict__ X,
                                                  const bf16_t* __restrict__ W1t,
                                                  const float* __restrict__ b1,
                                                  bf16_t* __restrict__ Hb,
                                                  int K, int N) {
  v8f acc[4][2];
  const v8f vz = {0.f,0.f,0.f,0.f,0.f,0.f,0.f,0.f};
  #pragma unroll
  for (int i = 0; i < 4; ++i)
    #pragma unroll
    for (int j = 0; j < 2; ++j) acc[i][j] = vz;

  gemm_tiles(X, W1t, K, blockIdx.y, blockIdx.x, acc);

  const int lane  = threadIdx.x & 31;
  const int wid   = threadIdx.x >> 5;
  const int mBase = blockIdx.y * TILE_M + (wid & 1) * 64 + ((lane >> 4) << 3);
  const int nBase = blockIdx.x * TILE_N + (wid >> 1) * 32 + (lane & 15);
  #pragma unroll
  for (int fm = 0; fm < 4; ++fm)
    #pragma unroll
    for (int fn = 0; fn < 2; ++fn) {
      const int n = nBase + fn * 16;
      const float bias = b1[n];
      #pragma unroll
      for (int r = 0; r < 8; ++r) {
        const int m = mBase + fm * 16 + r;
        const float v = acc[fm][fn][r] + bias;
        const float g = 0.5f * v * (1.0f + erff(v * 0.70710678118654752f));
        Hb[(size_t)m * N + n] = (bf16_t)g;
      }
    }
}

// GEMM2: Y = (H @ W2 + b2) * p7[m], output fp32 [T][H]
__global__ __launch_bounds__(256) void gemm2_scale(const bf16_t* __restrict__ Hb,
                                                   const bf16_t* __restrict__ W2t,
                                                   const float* __restrict__ b2,
                                                   const float* __restrict__ p7,
                                                   float* __restrict__ out,
                                                   int K, int N) {
  v8f acc[4][2];
  const v8f vz = {0.f,0.f,0.f,0.f,0.f,0.f,0.f,0.f};
  #pragma unroll
  for (int i = 0; i < 4; ++i)
    #pragma unroll
    for (int j = 0; j < 2; ++j) acc[i][j] = vz;

  gemm_tiles(Hb, W2t, K, blockIdx.y, blockIdx.x, acc);

  const int lane  = threadIdx.x & 31;
  const int wid   = threadIdx.x >> 5;
  const int mBase = blockIdx.y * TILE_M + (wid & 1) * 64 + ((lane >> 4) << 3);
  const int nBase = blockIdx.x * TILE_N + (wid >> 1) * 32 + (lane & 15);
  #pragma unroll
  for (int fm = 0; fm < 4; ++fm)
    #pragma unroll
    for (int r = 0; r < 8; ++r) {
      const int m = mBase + fm * 16 + r;
      const float pv = p7[m];
      #pragma unroll
      for (int fn = 0; fn < 2; ++fn) {
        const int n = nBase + fn * 16;
        out[(size_t)m * N + n] = (acc[fm][fn][r] + b2[n]) * pv;
      }
    }
}

// ---------------- launch ----------------

extern "C" void kernel_launch(void* const* d_in, const int* in_sizes, int n_in,
                              void* d_out, int out_size, void* d_ws, size_t ws_size,
                              hipStream_t stream) {
  const float* x  = (const float*)d_in[0];
  const float* Wr = (const float*)d_in[1];
  const float* W1 = (const float*)d_in[2];
  const float* b1 = (const float*)d_in[3];
  const float* W2 = (const float*)d_in[4];
  const float* b2 = (const float*)d_in[5];
  float* out = (float*)d_out;

  const int E = 8;
  const int H = in_sizes[1] / E;   // 1024
  const int F = in_sizes[3] / E;   // 4096
  const int T = in_sizes[0] / H;   // 16384
  const int eid = E - 1;

  uint8_t* ws = (uint8_t*)d_ws;
  size_t off = 0;
  bf16_t* xb  = (bf16_t*)(ws + off); off += (size_t)T * H * 2;
  bf16_t* w1t = (bf16_t*)(ws + off); off += (size_t)H * F * 2;
  bf16_t* w2t = (bf16_t*)(ws + off); off += (size_t)F * H * 2;
  float*  p7  = (float*) (ws + off); off += (size_t)T * 4;
  bf16_t* Hb  = (bf16_t*)(ws + off); off += (size_t)T * F * 2;

  // 1) x -> bf16
  cvt_f32_bf16<<<(T * H / 4 + 255) / 256, 256, 0, stream>>>(x, xb, T * H / 4);
  // 2,3) W1[7] -> [F][H] bf16 ; W2[7] -> [H][F] bf16
  dim3 tb(32, 8);
  transpose_cvt<<<dim3(F / 32, H / 32), tb, 0, stream>>>(W1 + (size_t)eid * H * F, w1t, H, F);
  transpose_cvt<<<dim3(H / 32, F / 32), tb, 0, stream>>>(W2 + (size_t)eid * F * H, w2t, F, H);
  // 4) router prob of expert 7
  router_p7<<<T / 8, 256, 0, stream>>>(x, Wr, p7, H);
  // 5) H = gelu(X @ W1 + b1)
  gemm1_gelu<<<dim3(F / TILE_N, T / TILE_M), 256, SMEM_BYTES, stream>>>(
      xb, w1t, b1 + (size_t)eid * F, Hb, H, F);
  // 6) out = (H @ W2 + b2) * p7
  gemm2_scale<<<dim3(H / TILE_N, T / TILE_M), 256, SMEM_BYTES, stream>>>(
      Hb, w2t, b2 + (size_t)eid * H, p7, out, F, H);
}
